// G_adapter_80547816670026
// MI455X (gfx1250) — matrix operand order repaired and checked by reading, not compiled, not run on hardware
//
#include <hip/hip_runtime.h>
#include <hip/hip_bf16.h>

// ---------------- problem constants ----------------
#define BB   64
#define NN   1024
#define DDIM 768
#define CC   8
#define TOK  (BB*NN)          // 65536 tokens
#define TILES (TOK/16)        // 4096 16-token tiles

// ---------------- layout constants -----------------
#define WD_P 776              // W_down f16 row pitch (halfs), [16 rows][776]
#define BU_P 24               // W_up' f16 row pitch (halfs), [768 rows][24] per dir
#define XS_P 72               // staged x chunk row pitch (halfs), [16 tok][72]

// small-weight block (float offsets)
#define SM_W1A 0
#define SM_W1B 64
#define SM_W2J 128
#define SM_B1J 192
#define SM_B2J 200
#define SM_BDN 208
#define SM_DIR 216
#define DIR_SZ 232
#define DW_WQ 0
#define DW_WK 64
#define DW_WV 128
#define DW_BQ 192
#define DW_BK 200
#define DW_BV 208
#define DW_CK 216
#define DW_CV 224
#define SM_FLOATS (SM_DIR + 2*DIR_SZ)   // 680

// workspace / LDS byte offsets (identical layouts -> single block copy)
#define OFF_WD   0                         // 16*776 halfs        = 24832 B
#define OFF_BU   24832                     // 2*768*24 halfs      = 73728 B
#define OFF_BUPP 98560                     // 2*768 floats        = 6144 B
#define OFF_SM   104704                    // 680 floats          = 2720 B
#define OFF_PW   107424                    // per-wave regions start
#define PW_SZ    6144                      // xs 4608 + xdown 1024 + o16 512
#define SMEM_TOTAL (OFF_PW + 8*PW_SZ)      // 156576 B dynamic LDS

typedef _Float16 h4   __attribute__((ext_vector_type(4)));
typedef _Float16 h8   __attribute__((ext_vector_type(8)));
typedef _Float16 v16h __attribute__((ext_vector_type(16)));
typedef float    v8f  __attribute__((ext_vector_type(8)));

static __device__ __forceinline__ v16h cat16(h8 a, h8 b) {
  return __builtin_shufflevector(a, b, 0,1,2,3,4,5,6,7,8,9,10,11,12,13,14,15);
}

// ---------------------------------------------------------------------------
// Prep kernel: fold Wo into W_up, fold noise into biases, emit f16 copies of
// W_down / W_up' in WMMA-B-friendly layouts. Single block, 256 threads.
// ---------------------------------------------------------------------------
__global__ __launch_bounds__(256) void adapter_prep(
    const float* __restrict__ Wd,  const float* __restrict__ bdn,
    const float* __restrict__ Wup, const float* __restrict__ bup,
    const float* __restrict__ Wi0, const float* __restrict__ bi0,
    const float* __restrict__ Wo0, const float* __restrict__ bo0,
    const float* __restrict__ Wi1, const float* __restrict__ bi1,
    const float* __restrict__ Wo1, const float* __restrict__ bo1,
    const float* __restrict__ W1j, const float* __restrict__ b1j,
    const float* __restrict__ W2j, const float* __restrict__ b2j,
    const float* __restrict__ kn,  const float* __restrict__ vn,
    char* __restrict__ ws)
{
  const int tid = threadIdx.x;
  _Float16* wd16 = (_Float16*)(ws + OFF_WD);
  _Float16* bu16 = (_Float16*)(ws + OFF_BU);
  float*    bupp = (float*)(ws + OFF_BUPP);
  float*    sm   = (float*)(ws + OFF_SM);

  // W_down^T as B-matrix rows: row n (=output channel, padded to 16) holds W_down[n][k]
  for (int i = tid; i < 16*WD_P; i += 256) {
    int n = i / WD_P, k = i % WD_P;
    wd16[i] = (n < CC && k < DDIM) ? (_Float16)Wd[n*DDIM + k] : (_Float16)0.0f;
  }

  for (int d = 0; d < 2; ++d) {
    const float* Wo = d ? Wo1 : Wo0;  const float* bo = d ? bo1 : bo0;
    const float* Wi = d ? Wi1 : Wi0;  const float* bi = d ? bi1 : bi0;

    // W_up' = W_up @ Wo : row n holds K values (k<8 valid, rest zero-pad)
    for (int i = tid; i < DDIM*BU_P; i += 256) {
      int n = i / BU_P, k = i % BU_P;
      float v = 0.f;
      if (k < CC) for (int c = 0; c < CC; ++c) v += Wup[n*CC + c] * Wo[c*CC + k];
      bu16[d*DDIM*BU_P + i] = (_Float16)v;
    }
    // b_up' = b_up + W_up @ bo
    for (int n = tid; n < DDIM; n += 256) {
      float v = bup[n];
      for (int c = 0; c < CC; ++c) v += Wup[n*CC + c] * bo[c];
      bupp[d*DDIM + n] = v;
    }
    float* dw = sm + SM_DIR + d*DIR_SZ;
    for (int i = tid; i < 192; i += 256) dw[i] = Wi[i];   // Wq | Wk | Wv
    if (tid < CC) {
      dw[DW_BQ + tid] = bi[tid];
      dw[DW_BK + tid] = bi[CC + tid];
      dw[DW_BV + tid] = bi[2*CC + tid];
      float ck = bi[CC + tid], cv = bi[2*CC + tid];
      for (int j = 0; j < CC; ++j) {
        ck += Wi[64  + tid*CC + j] * kn[d*CC + j];
        cv += Wi[128 + tid*CC + j] * vn[d*CC + j];
      }
      dw[DW_CK + tid] = ck;
      dw[DW_CV + tid] = cv;
    }
  }
  if (tid < 64) {
    int c = tid >> 3, j = tid & 7;
    sm[SM_W1A + tid] = W1j[c*16 + j];
    sm[SM_W1B + tid] = W1j[c*16 + 8 + j];
    sm[SM_W2J + tid] = W2j[tid];
  }
  if (tid < CC) {
    sm[SM_B1J + tid] = b1j[tid];
    sm[SM_B2J + tid] = b2j[tid];
    sm[SM_BDN + tid] = bdn[tid];
  }
}

// ---------------------------------------------------------------------------
// Fused main kernel. 256 threads = 8 waves; each wave owns one 16-token tile.
// ---------------------------------------------------------------------------
__global__ __launch_bounds__(256) void adapter_main(
    const float* __restrict__ x, const float* __restrict__ xi,
    const char* __restrict__ ws, float* __restrict__ out)
{
  extern __shared__ char smem[];

  // cooperative weight copy (layout identical to workspace)
  {
    const uint4* s = (const uint4*)ws;
    uint4* d = (uint4*)smem;
    for (int i = threadIdx.x; i < OFF_PW/16; i += 256) d[i] = s[i];
  }
  __syncthreads();

  const _Float16* sWd   = (const _Float16*)(smem + OFF_WD);
  const _Float16* sBu   = (const _Float16*)(smem + OFF_BU);
  const float*    sBupp = (const float*)(smem + OFF_BUPP);
  const float*    sm    = (const float*)(smem + OFF_SM);

  const int lane = threadIdx.x & 31;
  const int wave = threadIdx.x >> 5;
  char* pw = smem + OFF_PW + wave*PW_SZ;
  _Float16* sXs = (_Float16*)pw;                    // [2][16][XS_P] f16 staged chunks
  float*    sXd = (float*)(pw + 4608);              // [2][16][8]  f32 x_down/xi_down
  _Float16* sO  = (_Float16*)(pw + 4608 + 1024);    // [2][16][8]  f16 attention out

  const int tile = blockIdx.x*8 + wave;
  const long tb  = (long)tile * 16;
  const float* g0 = x  + tb*DDIM;
  const float* g1 = xi + tb*DDIM;

  const int mrow = lane & 15;     // token row (A/D M-index), or B column (N-index)
  const int hiK  = lane >> 4;     // lane half selector per WMMA register layouts

  // -------- down projection: [16,768] x [768,8->16] via 16x16x32 f16 WMMA --------
  v8f accx = {}; v8f acci = {};
  for (int kb = 0; kb < DDIM; kb += 64) {
    #pragma unroll
    for (int inp = 0; inp < 2; ++inp) {
      const float* src = inp ? g1 : g0;
      _Float16* dst = sXs + inp*(16*XS_P);
      #pragma unroll
      for (int i = 0; i < 8; ++i) {
        int e = (i*32 + lane)*4;            // 1024 floats per (input,chunk)
        int t = e >> 6, k = e & 63;
        float4 v = *(const float4*)(src + t*DDIM + kb + k);
        if (kb + 64 < DDIM) __builtin_prefetch(src + t*DDIM + kb + 64 + k, 0, 1);
        h4 hv; hv[0]=(_Float16)v.x; hv[1]=(_Float16)v.y; hv[2]=(_Float16)v.z; hv[3]=(_Float16)v.w;
        *(h4*)(dst + t*XS_P + k) = hv;
      }
    }
    __syncthreads();
    #pragma unroll
    for (int kk = 0; kk < 64; kk += 32) {
      // B fragment (32x16): lane holds k = base..base+7 and base+16..base+23 for column n=mrow
      int bcol = kb + kk + hiK*8;
      h8 b0 = *(const h8*)(sWd + mrow*WD_P + bcol);
      h8 b1 = *(const h8*)(sWd + mrow*WD_P + bcol + 16);
      v16h bf = cat16(b0, b1);
      // A fragment (16x32): lane holds k = base..base+7 and base+16..base+23
      int acol = kk + hiK*8;
      const _Float16* r0 = sXs + mrow*XS_P + acol;
      h8 a0 = *(const h8*)(r0);
      h8 a1 = *(const h8*)(r0 + 16);
      v16h af = cat16(a0, a1);
      accx = __builtin_amdgcn_wmma_f32_16x16x32_f16(false, af, false, bf, (short)0, accx, false, false);
      const _Float16* r1 = r0 + 16*XS_P;
      h8 c0 = *(const h8*)(r1);
      h8 c1 = *(const h8*)(r1 + 16);
      v16h cf = cat16(c0, c1);
      acci = __builtin_amdgcn_wmma_f32_16x16x32_f16(false, cf, false, bf, (short)0, acci, false, false);
    }
    __syncthreads();
  }

  // D-fragment -> LDS per-token layout, + b_down
  if (mrow < CC) {
    float bd = sm[SM_BDN + mrow];
    #pragma unroll
    for (int v = 0; v < 8; ++v) {
      int m = v + hiK*8;
      sXd[0*128 + m*8 + mrow] = accx[v] + bd;
      sXd[1*128 + m*8 + mrow] = acci[v] + bd;
    }
  }
  __syncthreads();

  // -------- middle phase: lane -> (token = mrow, direction = hiK); hd = 1 --------
  {
    const int d = hiK, t = mrow;
    float a[8], b[8];
    #pragma unroll
    for (int j = 0; j < 8; ++j) {
      a[j] = sXd[d*128 + t*8 + j];        // own down-projection (query side)
      b[j] = sXd[(1^d)*128 + t*8 + j];    // other stream
    }
    // judger: gelu(W1a a + W1b b + b1j) -> softmax(W2j h + b2j)
    float h_[8];
    #pragma unroll
    for (int c = 0; c < 8; ++c) {
      float tv = sm[SM_B1J + c];
      #pragma unroll
      for (int j = 0; j < 8; ++j)
        tv += sm[SM_W1A + c*8 + j]*a[j] + sm[SM_W1B + c*8 + j]*b[j];
      h_[c] = 0.5f*tv*(1.0f + erff(tv*0.70710678118654752f));  // exact gelu
    }
    float r[8], mx = -1e30f;
    #pragma unroll
    for (int c = 0; c < 8; ++c) {
      float rv = sm[SM_B2J + c];
      #pragma unroll
      for (int j = 0; j < 8; ++j) rv += sm[SM_W2J + c*8 + j]*h_[j];
      r[c] = rv; mx = fmaxf(mx, rv);
    }
    float ssum = 0.f;
    #pragma unroll
    for (int c = 0; c < 8; ++c) { r[c] = expf(r[c]-mx); ssum += r[c]; }
    float rinv = 1.0f/ssum;
    #pragma unroll
    for (int c = 0; c < 8; ++c) r[c] *= rinv;

    const float* dw = sm + SM_DIR + d*DIR_SZ;
    #pragma unroll
    for (int c = 0; c < 8; ++c) {   // per head (head_dim == 1)
      float q  = dw[DW_BQ+c], k0 = dw[DW_CK+c], k1 = dw[DW_BK+c];
      float v0 = dw[DW_CV+c], v1 = dw[DW_BV+c];
      #pragma unroll
      for (int j = 0; j < 8; ++j) {
        float wq = dw[DW_WQ + c*8 + j];
        float wk = dw[DW_WK + c*8 + j];
        float wv = dw[DW_WV + c*8 + j];
        q  += wq*a[j];
        k0 += wk*a[j];
        k1 += wk*(a[j]*r[j]);
        v0 += wv*a[j];
        v1 += wv*b[j];
      }
      float s0 = q*k0, s1 = q*k1;          // scale = 1/sqrt(1)
      float m2 = fmaxf(s0, s1);
      float e0 = expf(s0-m2), e1 = expf(s1-m2);
      float o = (e0*v0 + e1*v1)/(e0+e1);
      sO[d*128 + t*8 + c] = (_Float16)o;   // Wo folded into up-projection
    }
  }
  __syncthreads();

  // -------- up projection: [16,8->32] x [8->32,768] via WMMA, bias, store --------
  const h8 zz = {};
  #pragma unroll
  for (int d = 0; d < 2; ++d) {
    h8 lo = zz;
    if (lane < 16) lo = *(const h8*)(sO + d*128 + lane*8);   // k=0..7, rest zero
    v16h af = cat16(lo, zz);
    float* outd = out + (long)d*((long)TOK*DDIM) + tb*DDIM;
    const _Float16* bud = sBu + d*(DDIM*BU_P);
    const float* bpd = sBupp + d*DDIM;
    for (int nt = 0; nt < 48; ++nt) {
      h8 blo = zz, bhi = zz;
      if (lane < 16) {                       // lanes 16..31 cover k=16..31 -> zero
        const _Float16* br = bud + (nt*16 + mrow)*BU_P;
        blo = *(const h8*)br;                // k=0..7 (data)
        bhi = *(const h8*)(br + 8);          // k=8..15 (stored zeros)
      }
      v16h bf = cat16(blo, bhi);
      v8f dd = {};
      dd = __builtin_amdgcn_wmma_f32_16x16x32_f16(false, af, false, bf, (short)0, dd, false, false);
      float bias = bpd[nt*16 + mrow];
      #pragma unroll
      for (int v = 0; v < 8; ++v)
        outd[(long)(v + hiK*8)*DDIM + nt*16 + mrow] = dd[v] + bias;
    }
  }
}

// ---------------------------------------------------------------------------
extern "C" void kernel_launch(void* const* d_in, const int* in_sizes, int n_in,
                              void* d_out, int out_size, void* d_ws, size_t ws_size,
                              hipStream_t stream) {
  const float* x    = (const float*)d_in[0];
  const float* xi   = (const float*)d_in[1];
  const float* Wd   = (const float*)d_in[2];
  const float* bdn  = (const float*)d_in[3];
  const float* Wup  = (const float*)d_in[4];
  const float* bup  = (const float*)d_in[5];
  const float* Wi0  = (const float*)d_in[6];
  const float* bi0  = (const float*)d_in[7];
  const float* Wo0  = (const float*)d_in[8];
  const float* bo0  = (const float*)d_in[9];
  const float* Wi1  = (const float*)d_in[10];
  const float* bi1  = (const float*)d_in[11];
  const float* Wo1  = (const float*)d_in[12];
  const float* bo1  = (const float*)d_in[13];
  const float* W1j  = (const float*)d_in[14];
  const float* b1j  = (const float*)d_in[15];
  const float* W2j  = (const float*)d_in[16];
  const float* b2j  = (const float*)d_in[17];
  const float* kn   = (const float*)d_in[18];
  const float* vn   = (const float*)d_in[19];
  char* ws = (char*)d_ws;

  adapter_prep<<<1, 256, 0, stream>>>(Wd, bdn, Wup, bup, Wi0, bi0, Wo0, bo0,
                                      Wi1, bi1, Wo1, bo1, W1j, b1j, W2j, b2j,
                                      kn, vn, ws);

  (void)hipFuncSetAttribute((const void*)adapter_main,
                            hipFuncAttributeMaxDynamicSharedMemorySize, SMEM_TOTAL);
  adapter_main<<<dim3(TILES/8), dim3(256), SMEM_TOTAL, stream>>>(x, xi, ws, (float*)d_out);
}
